// GINEncoder_6640019439960
// MI455X (gfx1250) — matrix operand order repaired and checked by reading, not compiled, not run on hardware
//
#include <hip/hip_runtime.h>
#include <hip/hip_bf16.h>
#include <math.h>

typedef __attribute__((ext_vector_type(16))) __bf16 v16bf;
typedef __attribute__((ext_vector_type(8)))  float  v8f;

#define N_NODES 100000
#define N_EDGES 640000
#define HID     128
#define NLAYERS 3
#define NGROUPS 512
#define FTOT    (NLAYERS*HID)   /* 384 */
#define BN_EPS  1e-5f
#define LDW     136             /* padded LDS row stride (bf16 elements) */

// ---------------- kernel: agg = h (residual/self term of GIN) ----------------
__global__ void copy_kernel(const float* __restrict__ h, float* __restrict__ agg) {
    int idx = blockIdx.x * blockDim.x + threadIdx.x;        // over N*32 float4
    if (idx >= N_NODES * 32) return;
    ((float4*)agg)[idx] = ((const float4*)h)[idx];
}

// ---------------- kernel: agg[dst] += h[src], one wave per edge --------------
__global__ void scatter_kernel(const float* __restrict__ h,
                               const int* __restrict__ src,
                               const int* __restrict__ dst,
                               float* __restrict__ agg) {
    long long t = (long long)blockIdx.x * blockDim.x + threadIdx.x;
    int e    = (int)(t >> 5);
    int lane = (int)(t & 31);
    if (e >= N_EDGES) return;
    int s = src[e], d = dst[e];
    float4 v = ((const float4*)(h + (long long)s * HID))[lane];
    float* a = agg + (long long)d * HID + lane * 4;
    atomicAdd(a + 0, v.x); atomicAdd(a + 1, v.y);
    atomicAdd(a + 2, v.z); atomicAdd(a + 3, v.w);
}

// ------------- fused MLP: elu(relu(X@W1+b1)@W2+b2) + BN partial sums ---------
__launch_bounds__(256)
__global__ void mlp_kernel(const float* __restrict__ X,      // [N,128] aggregated
                           const float* __restrict__ W1,     // [128,128] (k-major)
                           const float* __restrict__ b1,
                           const float* __restrict__ W2,
                           const float* __restrict__ b2,
                           float* __restrict__ out_xs,       // [N,FTOT], col_off slice
                           int col_off,
                           float* __restrict__ stat_sum,     // [128]
                           float* __restrict__ stat_sq) {    // [128]
    __shared__ __bf16 sW1[HID * LDW];       // transposed: [n][k]
    __shared__ __bf16 sW2[HID * LDW];       // transposed: [n][k]
    __shared__ __bf16 sXY[HID * LDW];       // X tile, reused as ReLU(Y1) tile
    __shared__ float  sB1[HID], sB2[HID];
    __shared__ float  sSum[HID], sSq[HID];

    const int tid  = threadIdx.x;
    const int lane = tid & 31;
    const int wave = tid >> 5;
    const int row0 = blockIdx.x * HID;      // 128 rows per block

    // cooperative weight load (transpose k-major -> n-major, f32 -> bf16)
    for (int j = 0; j < 64; ++j) {
        int idx = j * 256 + tid;            // 0..16383
        int k = idx >> 7, n = idx & 127;
        sW1[n * LDW + k] = (__bf16)W1[idx];
        sW2[n * LDW + k] = (__bf16)W2[idx];
    }
    if (tid < HID) { sB1[tid] = b1[tid]; sB2[tid] = b2[tid]; sSum[tid] = 0.f; sSq[tid] = 0.f; }

    // cooperative X tile load (f32 -> bf16), zero-fill past N
    for (int j = 0; j < 16; ++j) {
        int idx = j * 256 + tid;            // float4 index 0..4095
        int r = idx >> 5, c4 = idx & 31;
        float4 v = make_float4(0.f, 0.f, 0.f, 0.f);
        if (row0 + r < N_NODES)
            v = ((const float4*)(X + (long long)(row0 + r) * HID))[c4];
        __bf16* p = &sXY[r * LDW + c4 * 4];
        p[0] = (__bf16)v.x; p[1] = (__bf16)v.y; p[2] = (__bf16)v.z; p[3] = (__bf16)v.w;
    }
    __syncthreads();

    const int m0   = wave * 16;             // wave-private 16-row slab
    const int nsel = lane & 15;
    const int hi   = lane >> 4;             // 0 or 1 (K-half select)

    // A fragment (16x32 bf16): lane = row, K half 0-15 / 16-31 selected by hi
    auto loadA = [&](const __bf16* base, int k0) -> v16bf {
        v16bf a;
        const __bf16* p0 = base + (m0 + nsel) * LDW + k0 + hi * 16;
        #pragma unroll
        for (int e = 0; e < 8; ++e) a[e] = p0[e];           // K = k0+hi*16 + 0..7
        #pragma unroll
        for (int e = 0; e < 8; ++e) a[8 + e] = p0[8 + e];   // K = +8..+15
        return a;
    };
    // B fragment (32x16 bf16): lane = column (n), K half 0-15 / 16-31 by hi
    auto loadB = [&](const __bf16* base, int n0, int k0) -> v16bf {
        v16bf b;
        const __bf16* p0 = base + (n0 + nsel) * LDW + k0 + hi * 16;
        #pragma unroll
        for (int e = 0; e < 16; ++e) b[e] = p0[e];
        return b;
    };

    const v8f zero = {0.f, 0.f, 0.f, 0.f, 0.f, 0.f, 0.f, 0.f};
    v8f acc[8];

    // ---------------- stage 1: Y = relu(X @ W1 + b1) ----------------
    #pragma unroll
    for (int t = 0; t < 8; ++t) acc[t] = zero;
    for (int k0 = 0; k0 < HID; k0 += 32) {
        v16bf a = loadA(sXY, k0);
        #pragma unroll
        for (int t = 0; t < 8; ++t) {
            v16bf b = loadB(sW1, t * 16, k0);
            acc[t] = __builtin_amdgcn_wmma_f32_16x16x32_bf16(
                false, a, false, b, (short)0, acc[t], false, false);
        }
    }
    // epilogue into wave-private LDS rows (no cross-wave hazard)
    #pragma unroll
    for (int t = 0; t < 8; ++t) {
        int n = t * 16 + nsel;
        float bias = sB1[n];
        #pragma unroll
        for (int r = 0; r < 8; ++r) {
            float y = acc[t][r] + bias;
            y = y > 0.f ? y : 0.f;
            sXY[(m0 + r + hi * 8) * LDW + n] = (__bf16)y;
        }
    }
    __syncthreads();

    // ---------------- stage 2: Z = elu(Y @ W2 + b2) ----------------
    #pragma unroll
    for (int t = 0; t < 8; ++t) acc[t] = zero;
    for (int k0 = 0; k0 < HID; k0 += 32) {
        v16bf a = loadA(sXY, k0);
        #pragma unroll
        for (int t = 0; t < 8; ++t) {
            v16bf b = loadB(sW2, t * 16, k0);
            acc[t] = __builtin_amdgcn_wmma_f32_16x16x32_bf16(
                false, a, false, b, (short)0, acc[t], false, false);
        }
    }
    #pragma unroll
    for (int t = 0; t < 8; ++t) {
        int n = t * 16 + nsel;          // all 8 regs of this tile share column n
        float bias = sB2[n];
        float csum = 0.f, csq = 0.f;
        #pragma unroll
        for (int r = 0; r < 8; ++r) {
            int row = row0 + m0 + r + hi * 8;
            float y = acc[t][r] + bias;
            y = y > 0.f ? y : (__expf(y) - 1.f);            // elu, alpha=1
            if (row < N_NODES) {
                out_xs[(long long)row * FTOT + col_off + n] = y;
                csum += y; csq += y * y;
            }
        }
        atomicAdd(&sSum[n], csum);
        atomicAdd(&sSq[n],  csq);
    }
    __syncthreads();
    if (tid < HID) {
        atomicAdd(&stat_sum[tid], sSum[tid]);
        atomicAdd(&stat_sq[tid],  sSq[tid]);
    }
}

// ---------------- BN: per-column scale/shift from batch stats ----------------
__global__ void bn_finalize_kernel(const float* __restrict__ sum,
                                   const float* __restrict__ sq,
                                   const float* __restrict__ gamma,
                                   const float* __restrict__ beta,
                                   float* __restrict__ scale,
                                   float* __restrict__ shift) {
    int c = threadIdx.x;
    if (c >= HID) return;
    const float invN = 1.f / (float)N_NODES;
    float mu  = sum[c] * invN;
    float var = sq[c] * invN - mu * mu;
    float sc  = rsqrtf(var + BN_EPS) * gamma[c];
    scale[c] = sc;
    shift[c] = beta[c] - mu * sc;
}

// --------- normalize in place, stash h for next layer, pooled scatter --------
__global__ void norm_pool_kernel(float* __restrict__ out_xs, int col_off,
                                 const float* __restrict__ scale,
                                 const float* __restrict__ shift,
                                 const int* __restrict__ batch,
                                 float* __restrict__ hbuf,
                                 float* __restrict__ pooled) {
    int idx = blockIdx.x * blockDim.x + threadIdx.x;        // over N*32 float4
    if (idx >= N_NODES * 32) return;
    int row = idx >> 5, c4 = (idx & 31) * 4;
    float* p = out_xs + (long long)row * FTOT + col_off + c4;
    float4 v  = *(float4*)p;
    float4 sc = *(const float4*)(scale + c4);
    float4 sh = *(const float4*)(shift + c4);
    v.x = v.x * sc.x + sh.x;
    v.y = v.y * sc.y + sh.y;
    v.z = v.z * sc.z + sh.z;
    v.w = v.w * sc.w + sh.w;
    *(float4*)p = v;
    *(float4*)(hbuf + (long long)row * HID + c4) = v;
    int g = batch[row];
    float* pp = pooled + (long long)g * FTOT + col_off + c4;
    atomicAdd(pp + 0, v.x); atomicAdd(pp + 1, v.y);
    atomicAdd(pp + 2, v.z); atomicAdd(pp + 3, v.w);
}

extern "C" void kernel_launch(void* const* d_in, const int* in_sizes, int n_in,
                              void* d_out, int out_size, void* d_ws, size_t ws_size,
                              hipStream_t stream) {
    const float* x     = (const float*)d_in[0];
    const int*   edge  = (const int*)d_in[1];
    const int*   src   = edge;
    const int*   dst   = edge + N_EDGES;
    const int*   batch = (const int*)d_in[2];
    const float* W1    = (const float*)d_in[3];
    const float* b1    = (const float*)d_in[4];
    const float* W2    = (const float*)d_in[5];
    const float* b2    = (const float*)d_in[6];
    const float* gamma = (const float*)d_in[7];
    const float* beta  = (const float*)d_in[8];

    float* pooled = (float*)d_out;                               // [G, FTOT]
    float* xs     = (float*)d_out + (size_t)NGROUPS * FTOT;      // [N, FTOT]

    float* ws     = (float*)d_ws;
    float* hbuf   = ws;                                          // N*HID
    float* agg    = ws + (size_t)N_NODES * HID;                  // N*HID
    float* stat   = agg + (size_t)N_NODES * HID;                 // 512 floats
    float* ssum   = stat;
    float* ssq    = stat + 128;
    float* sscale = stat + 256;
    float* sshift = stat + 384;

    hipMemsetAsync(pooled, 0, (size_t)NGROUPS * FTOT * sizeof(float), stream);

    const int copyBlocks = (N_NODES * 32 + 255) / 256;
    const int scatBlocks = (int)(((long long)N_EDGES * 32 + 255) / 256);
    const int mlpBlocks  = (N_NODES + 127) / 128;

    for (int l = 0; l < NLAYERS; ++l) {
        const float* hin = (l == 0) ? x : hbuf;
        copy_kernel   <<<copyBlocks, 256, 0, stream>>>(hin, agg);
        scatter_kernel<<<scatBlocks, 256, 0, stream>>>(hin, src, dst, agg);
        hipMemsetAsync(stat, 0, 256 * sizeof(float), stream);
        mlp_kernel    <<<mlpBlocks, 256, 0, stream>>>(agg,
                                                      W1 + (size_t)l * HID * HID, b1 + l * HID,
                                                      W2 + (size_t)l * HID * HID, b2 + l * HID,
                                                      xs, l * HID, ssum, ssq);
        bn_finalize_kernel<<<1, 128, 0, stream>>>(ssum, ssq, gamma + l * HID, beta + l * HID,
                                                  sscale, sshift);
        norm_pool_kernel<<<copyBlocks, 256, 0, stream>>>(xs, l * HID, sscale, sshift,
                                                         batch, hbuf, pooled);
    }
}
